// Attention_43181601194655
// MI455X (gfx1250) — matrix operand order repaired and checked by reading, not compiled, not run on hardware
//
#include <hip/hip_runtime.h>
#include <stdint.h>

// ---------------- problem constants ----------------
constexpr int kB    = 2;
constexpr int kS    = 2048;
constexpr int kHID  = 2048;
constexpr int kH    = 32;
constexpr int kHKV  = 8;
constexpr int kD    = 64;
constexpr int kM    = kB * kS;          // 4096 rows for projection GEMMs
constexpr float kScale = 0.125f;        // D^-0.5
constexpr float kL2E   = 1.4426950408889634f;

typedef __attribute__((ext_vector_type(16))) __bf16 v16bf;
typedef __attribute__((ext_vector_type(8)))  __bf16 v8bf;
typedef __attribute__((ext_vector_type(8)))  float  v8f;
typedef unsigned int u32x4 __attribute__((ext_vector_type(4)));
typedef int          i32x4 __attribute__((ext_vector_type(4)));
typedef int          i32x8 __attribute__((ext_vector_type(8)));

#define WMMA_BF16(a, b, c) \
  __builtin_amdgcn_wmma_f32_16x16x32_bf16(false, (a), false, (b), (short)0, (c), false, false)

#if defined(__AMDGCN__) && __has_builtin(__builtin_amdgcn_tensor_load_to_lds)
#define HAVE_TDM 1
#else
#define HAVE_TDM 0
#endif

static __device__ __forceinline__ v16bf combine8(v8bf lo, v8bf hi) {
  union { v16bf v; struct { v8bf lo, hi; } s; } u;
  u.s.lo = lo; u.s.hi = hi;
  return u.v;
}

#if HAVE_TDM
// Low 32 bits of a generic pointer into the LDS aperture == LDS byte offset.
static __device__ __forceinline__ uint32_t lds_off(const void* p) {
  return (uint32_t)(uintptr_t)p;
}

// TDM D# for a 128-row x 32-elem bf16 tile, row stride k_elems, destination
// rows padded 64B data + 32B pad (=> 96B LDS row stride, matches sX[128][48]).
static __device__ __forceinline__ void tdm_load_tile_128x32(
    const void* gptr, uint32_t lds_byte_off, uint32_t k_elems) {
  uint64_t ga = (uint64_t)(uintptr_t)gptr;
  u32x4 g0;
  g0[0] = 1u;                                    // count=1, user descriptor
  g0[1] = lds_byte_off;                          // lds_addr
  g0[2] = (uint32_t)ga;                          // global_addr[31:0]
  g0[3] = (uint32_t)((ga >> 32) & 0x01FFFFFFu)   // global_addr[56:32]
        | (2u << 30);                            // type = 2 ("image")
  i32x8 g1;
  g1[0] = (int)((1u << 16)    // data_size = 1 -> 2 bytes
              | (1u << 20)    // pad_enable
              | (3u << 22)    // pad_interval: pad after every 16 DWORDs (64B)
              | (7u << 25));  // pad_amount: 8 DWORDs (32B)
  g1[1] = (int)((k_elems & 0xFFFFu) << 16);          // tensor_dim0[15:0]
  g1[2] = (int)((k_elems >> 16) | (128u << 16));     // tensor_dim0[31:16] | tensor_dim1=128
  g1[3] = (int)(32u << 16);                          // tensor_dim1[31:16]=0 | tile_dim0=32
  g1[4] = (int)128u;                                 // tile_dim1=128, tile_dim2=0
  g1[5] = (int)k_elems;                              // tensor_dim0_stride[31:0]
  g1[6] = 0;                                         // stride0[47:32] | stride1[15:0]
  g1[7] = 0;                                         // stride1[47:16]
  i32x4 z4 = {0, 0, 0, 0};
#if __clang_major__ >= 23
  i32x8 z8 = {0, 0, 0, 0, 0, 0, 0, 0};
  __builtin_amdgcn_tensor_load_to_lds(g0, g1, z4, z4, z8, 0);
#else
  __builtin_amdgcn_tensor_load_to_lds(g0, g1, z4, z4, 0);
#endif
}
#endif  // HAVE_TDM

// Stage one 128x32 A tile and one 128x32 B tile into LDS buffers.
static __device__ __forceinline__ void stage_tiles(
    const __bf16* __restrict__ A, const __bf16* __restrict__ Bt, int K,
    int m0, int n0, int k0, __bf16 (*bufA)[48], __bf16 (*bufB)[48], int tid) {
#if HAVE_TDM
  if ((tid >> 5) == 0) {   // one wave issues the DMA; TENSORcnt tracks it
    tdm_load_tile_128x32(A  + (size_t)m0 * K + k0, lds_off(bufA), (uint32_t)K);
    tdm_load_tile_128x32(Bt + (size_t)n0 * K + k0, lds_off(bufB), (uint32_t)K);
  }
#else
  int r = tid >> 1, c = (tid & 1) * 16;
  *(v16bf*)&bufA[r][c] = *(const v16bf*)(A  + (size_t)(m0 + r) * K + k0 + c);
  *(v16bf*)&bufB[r][c] = *(const v16bf*)(Bt + (size_t)(n0 + r) * K + k0 + c);
#endif
}

static __device__ __forceinline__ void stage_wait(int tid) {
#if HAVE_TDM
  if ((tid >> 5) == 0) __builtin_amdgcn_s_wait_tensorcnt(0);
#endif
}

// ---------------- elementwise convert kernels ----------------
__global__ void cvt_bf16_kernel(const float* __restrict__ in,
                                __bf16* __restrict__ out, int n) {
  int i = blockIdx.x * blockDim.x + threadIdx.x;
  if (i < n) out[i] = (__bf16)in[i];
}

// out[n*K + k] = (bf16) in[k*N + n]   (weight transpose so GEMM is TN)
__global__ void transpose_bf16_kernel(const float* __restrict__ in,
                                      __bf16* __restrict__ out, int K, int N) {
  int i = blockIdx.x * blockDim.x + threadIdx.x;
  if (i < N * K) {
    int nn = i / K, kk = i - nn * K;
    out[i] = (__bf16)in[(size_t)kk * N + nn];
  }
}

// ---------------- RoPE kernels ----------------
// q_f32: [B*S, H*D] -> qbf: [B, H, S, D], rope applied, pre-scaled by kScale
__global__ void rope_q_kernel(const float* __restrict__ q,
                              const float* __restrict__ cosT,
                              const float* __restrict__ sinT,
                              __bf16* __restrict__ qbf) {
  int t = blockIdx.x * blockDim.x + threadIdx.x;   // over B*S*H*32
  if (t >= kB * kS * kH * 32) return;
  int d  = t & 31;
  int h  = (t >> 5) & (kH - 1);
  int bs = t >> 10;
  int s  = bs & (kS - 1);
  int b  = bs >> 11;
  const float* qp = q + (size_t)bs * (kH * kD) + h * kD;
  float v1 = qp[d], v2 = qp[d + 32];
  float c1 = cosT[s * kD + d],      sn1 = sinT[s * kD + d];
  float c2 = cosT[s * kD + d + 32], sn2 = sinT[s * kD + d + 32];
  __bf16* op = qbf + ((size_t)(b * kH + h) * kS + s) * kD;
  op[d]      = (__bf16)(kScale * (v1 * c1 - v2 * sn1));
  op[d + 32] = (__bf16)(kScale * (v2 * c2 + v1 * sn2));
}

// k_f32,v_f32: [B*S, HKV*D] -> kbf: [B,HKV,S,D] (rope), vt: [B,HKV,D,S]
__global__ void rope_kv_kernel(const float* __restrict__ k,
                               const float* __restrict__ v,
                               const float* __restrict__ cosT,
                               const float* __restrict__ sinT,
                               __bf16* __restrict__ kbf,
                               __bf16* __restrict__ vt) {
  int t = blockIdx.x * blockDim.x + threadIdx.x;   // over B*S*HKV*32
  if (t >= kB * kS * kHKV * 32) return;
  int d  = t & 31;
  int hk = (t >> 5) & (kHKV - 1);
  int bs = t >> 8;
  int s  = bs & (kS - 1);
  int b  = bs >> 11;
  const float* kp = k + (size_t)bs * (kHKV * kD) + hk * kD;
  const float* vp = v + (size_t)bs * (kHKV * kD) + hk * kD;
  float k1 = kp[d], k2 = kp[d + 32];
  float c1 = cosT[s * kD + d],      sn1 = sinT[s * kD + d];
  float c2 = cosT[s * kD + d + 32], sn2 = sinT[s * kD + d + 32];
  __bf16* ko = kbf + ((size_t)(b * kHKV + hk) * kS + s) * kD;
  ko[d]      = (__bf16)(k1 * c1 - k2 * sn1);
  ko[d + 32] = (__bf16)(k2 * c2 + k1 * sn2);
  size_t vtb = (size_t)(b * kHKV + hk) * kD;
  vt[(vtb + d) * kS + s]      = (__bf16)vp[d];
  vt[(vtb + d + 32) * kS + s] = (__bf16)vp[d + 32];
}

// ---------------- TDM double-buffered WMMA GEMM:  C = A[M,K] * Bt[N,K]^T ----
// block tile 128x128x32, 8 waves (4 along M x 2 along N), wave tile 32x64
__global__ __launch_bounds__(256) void gemm_bf16_tn(
    const __bf16* __restrict__ A, const __bf16* __restrict__ Bt,
    float* __restrict__ C, int M, int N, int K) {
  __shared__ __bf16 sA[2][128][48];   // 96B row stride (TDM pad writes it)
  __shared__ __bf16 sB[2][128][48];
  const int tid  = threadIdx.x;
  const int wid  = tid >> 5;
  const int lane = tid & 31;
  const int hl   = lane >> 4;
  const int l16  = lane & 15;
  const int m0 = blockIdx.y * 128, n0 = blockIdx.x * 128;
  const int wm = (wid >> 1) * 32;  // wave row offset (0,32,64,96)
  const int wn = (wid & 1) * 64;   // wave col offset (0,64)

  const v8f vzero = {0.f, 0.f, 0.f, 0.f, 0.f, 0.f, 0.f, 0.f};
  v8f acc[2][4];
#pragma unroll
  for (int i = 0; i < 2; ++i)
#pragma unroll
    for (int j = 0; j < 4; ++j) acc[i][j] = vzero;

  // prologue: stage first K-slab into buffer 0
  stage_tiles(A, Bt, K, m0, n0, 0, sA[0], sB[0], tid);
  stage_wait(tid);
  __syncthreads();

  int cur = 0;
  for (int k0 = 0; k0 < K; k0 += 32) {
    // async-stage next slab into the other buffer while we compute this one
    if (k0 + 32 < K)
      stage_tiles(A, Bt, K, m0, n0, k0 + 32, sA[cur ^ 1], sB[cur ^ 1], tid);

    v16bf af[2], bfr[4];
#pragma unroll
    for (int mi = 0; mi < 2; ++mi) {
      int ar = wm + mi * 16 + l16;   // A frag: M = lane&15, K split by half
      af[mi] = combine8(*(const v8bf*)&sA[cur][ar][16 * hl],
                        *(const v8bf*)&sA[cur][ar][16 * hl + 8]);
    }
#pragma unroll
    for (int ni = 0; ni < 4; ++ni) {
      int br = wn + ni * 16 + l16;   // B frag: N = lane&15, K = 16*half + j
      bfr[ni] = *(const v16bf*)&sB[cur][br][16 * hl];
    }
#pragma unroll
    for (int mi = 0; mi < 2; ++mi)
#pragma unroll
      for (int ni = 0; ni < 4; ++ni)
        acc[mi][ni] = WMMA_BF16(af[mi], bfr[ni], acc[mi][ni]);

    stage_wait(tid);      // TENSORcnt==0: next buffer fully landed in LDS
    __syncthreads();      // all waves done reading cur; DMA data visible
    cur ^= 1;
  }

#pragma unroll
  for (int mi = 0; mi < 2; ++mi)
#pragma unroll
    for (int ni = 0; ni < 4; ++ni)
#pragma unroll
      for (int r = 0; r < 8; ++r) {
        int row = m0 + wm + mi * 16 + r + 8 * hl;   // C/D: M = r + 8*half
        int col = n0 + wn + ni * 16 + l16;          //      N = lane&15
        C[(size_t)row * N + col] = acc[mi][ni][r];
      }
}

// ---------------- causal flash attention ----------------
// Qb: [B,H,S,D] bf16 (pre-scaled), Kb: [B,HKV,S,D] bf16, Vt: [B,HKV,D,S] bf16
// Ob: [B,S,H*D] bf16. One wave = 16 query rows; block = 8 waves = 128 rows.
__global__ __launch_bounds__(256) void flash_attn_kernel(
    const __bf16* __restrict__ Qb, const __bf16* __restrict__ Kb,
    const __bf16* __restrict__ Vt, __bf16* __restrict__ Ob) {
  __shared__ __bf16 sP[8][16][40];   // per-wave P-transpose buffer (80B rows)
  const int tid  = threadIdx.x;
  const int wid  = tid >> 5;
  const int lane = tid & 31;
  const int hl   = lane >> 4;
  const int l16  = lane & 15;
  const int b  = blockIdx.z;
  const int h  = blockIdx.y;
  const int hk = h >> 2;                         // GROUPS = H/HKV = 4
  const int qb = blockIdx.x * 128 + wid * 16;    // wave's first query row

  const __bf16* qrp = Qb + ((size_t)(b * kH + h) * kS + qb + l16) * kD;
  v16bf qf0 = combine8(*(const v8bf*)(qrp + 16 * hl),
                       *(const v8bf*)(qrp + 8 + 16 * hl));
  v16bf qf1 = combine8(*(const v8bf*)(qrp + 32 + 16 * hl),
                       *(const v8bf*)(qrp + 40 + 16 * hl));

  const __bf16* Kbase = Kb + (size_t)(b * kHKV + hk) * kS * kD;
  const __bf16* Vbase = Vt + (size_t)(b * kHKV + hk) * kD * kS;

  const v8f vzero = {0.f, 0.f, 0.f, 0.f, 0.f, 0.f, 0.f, 0.f};
  float m_r[8], l_r[8];
  v8f of[4];
#pragma unroll
  for (int r = 0; r < 8; ++r) { m_r[r] = -1e30f; l_r[r] = 0.f; }
#pragma unroll
  for (int n = 0; n < 4; ++n) of[n] = vzero;

  const int kend = qb + 16;                 // causal: keys < qb+16
  for (int kt = 0; kt < kend; kt += 32) {
    // ---- scores: two 16x16 tiles, each = 2 chained WMMAs over D=64 ----
    v8f s0a = vzero, s1a = vzero;
    {
      const __bf16* kp0 = Kbase + (size_t)(kt + l16) * kD;
      v16bf k00 = *(const v16bf*)(kp0 + 16 * hl);
      v16bf k01 = *(const v16bf*)(kp0 + 32 + 16 * hl);
      s0a = WMMA_BF16(qf0, k00, s0a);
      s0a = WMMA_BF16(qf1, k01, s0a);
      const __bf16* kp1 = Kbase + (size_t)(kt + 16 + l16) * kD;
      v16bf k10 = *(const v16bf*)(kp1 + 16 * hl);
      v16bf k11 = *(const v16bf*)(kp1 + 32 + 16 * hl);
      s1a = WMMA_BF16(qf0, k10, s1a);
      s1a = WMMA_BF16(qf1, k11, s1a);
    }

    // prefetch next key/value tiles into cache (global_prefetch_b8)
    if (kt + 32 < kend) {
      __builtin_prefetch(Kbase + (size_t)(kt + 32 + l16) * kD, 0, 0);
      __builtin_prefetch(Kbase + (size_t)(kt + 48 + l16) * kD, 0, 0);
      __builtin_prefetch(Vbase + (size_t)l16 * kS + kt + 32, 0, 0);
    }

    // ---- causal mask + online softmax (per row, butterfly within half) ----
#pragma unroll
    for (int r = 0; r < 8; ++r) {
      int qrow = qb + r + 8 * hl;
      float x0 = (kt + l16      <= qrow) ? s0a[r] : -1e30f;
      float x1 = (kt + 16 + l16 <= qrow) ? s1a[r] : -1e30f;
      float x = fmaxf(x0, x1);
#pragma unroll
      for (int off = 1; off < 16; off <<= 1) x = fmaxf(x, __shfl_xor(x, off, 32));
      float mnew  = fmaxf(m_r[r], x);
      float alpha = exp2f((m_r[r] - mnew) * kL2E);
      float p0 = exp2f((x0 - mnew) * kL2E);
      float p1 = exp2f((x1 - mnew) * kL2E);
      float rs = p0 + p1;
#pragma unroll
      for (int off = 1; off < 16; off <<= 1) rs += __shfl_xor(rs, off, 32);
      l_r[r] = l_r[r] * alpha + rs;
      m_r[r] = mnew;
#pragma unroll
      for (int n = 0; n < 4; ++n) of[n][r] *= alpha;
      sP[wid][r + 8 * hl][l16]      = (__bf16)p0;   // C-layout -> LDS
      sP[wid][r + 8 * hl][16 + l16] = (__bf16)p1;
    }

    // wave-private LDS transpose: C layout -> A layout (DS in-order per wave)
    __builtin_amdgcn_wave_barrier();
    v16bf pa = combine8(*(const v8bf*)&sP[wid][l16][16 * hl],
                        *(const v8bf*)&sP[wid][l16][16 * hl + 8]);
    __builtin_amdgcn_wave_barrier();

    // ---- P @ V: 4 WMMAs across D=64 ----
#pragma unroll
    for (int n = 0; n < 4; ++n) {
      const __bf16* vp = Vbase + (size_t)(n * 16 + l16) * kS + kt + 16 * hl;
      v16bf vf = *(const v16bf*)vp;
      of[n] = WMMA_BF16(pa, vf, of[n]);
    }
  }

  // ---- normalize and store [B,S,H*D] bf16 for the O-projection GEMM ----
#pragma unroll
  for (int n = 0; n < 4; ++n)
#pragma unroll
    for (int r = 0; r < 8; ++r) {
      int q = qb + r + 8 * hl;
      float inv = 1.f / l_r[r];
      Ob[((size_t)(b * kS + q) * kH + h) * kD + n * 16 + l16] =
          (__bf16)(of[n][r] * inv);
    }
}

// ---------------- host launcher ----------------
extern "C" void kernel_launch(void* const* d_in, const int* in_sizes, int n_in,
                              void* d_out, int out_size, void* d_ws, size_t ws_size,
                              hipStream_t stream) {
  const float* hidden = (const float*)d_in[0];
  const float* cosT   = (const float*)d_in[1];
  const float* sinT   = (const float*)d_in[2];
  // d_in[3] = attention_mask (causal; recomputed in-kernel)
  const float* Wq = (const float*)d_in[4];
  const float* Wk = (const float*)d_in[5];
  const float* Wv = (const float*)d_in[6];
  const float* Wo = (const float*)d_in[7];
  float* out = (float*)d_out;

  char* ws = (char*)d_ws;
  size_t off = 0;
  auto alloc = [&](size_t bytes) {
    char* p = ws + off;
    off = (off + bytes + 255) & ~(size_t)255;
    return p;
  };
  __bf16* Xbf  = (__bf16*)alloc((size_t)kM * kHID * 2);          // 16 MB
  __bf16* Wqt  = (__bf16*)alloc((size_t)(kH * kD) * kHID * 2);   // 8 MB
  __bf16* Wkt  = (__bf16*)alloc((size_t)(kHKV * kD) * kHID * 2); // 2 MB
  __bf16* Wvt  = (__bf16*)alloc((size_t)(kHKV * kD) * kHID * 2); // 2 MB
  __bf16* Wot  = (__bf16*)alloc((size_t)kHID * (kH * kD) * 2);   // 8 MB
  float*  qf32 = (float*)alloc((size_t)kM * kH * kD * 4);        // 32 MB
  float*  kf32 = (float*)alloc((size_t)kM * kHKV * kD * 4);      // 8 MB
  float*  vf32 = (float*)alloc((size_t)kM * kHKV * kD * 4);      // 8 MB
  __bf16* qbf  = (__bf16*)alloc((size_t)kM * kH * kD * 2);       // 16 MB
  __bf16* kbf  = (__bf16*)alloc((size_t)kM * kHKV * kD * 2);     // 4 MB
  __bf16* vtbf = (__bf16*)alloc((size_t)kM * kHKV * kD * 2);     // 4 MB
  __bf16* attnbf = (__bf16*)qf32;   // reuse: qf32 dead after RoPE

  const int T = 256;

  // 1) convert X, transpose-convert weights
  { int n = kM * kHID;
    cvt_bf16_kernel<<<(n + T - 1) / T, T, 0, stream>>>(hidden, Xbf, n); }
  { int n = (kH * kD) * kHID;
    transpose_bf16_kernel<<<(n + T - 1) / T, T, 0, stream>>>(Wq, Wqt, kHID, kH * kD); }
  { int n = (kHKV * kD) * kHID;
    transpose_bf16_kernel<<<(n + T - 1) / T, T, 0, stream>>>(Wk, Wkt, kHID, kHKV * kD); }
  { int n = (kHKV * kD) * kHID;
    transpose_bf16_kernel<<<(n + T - 1) / T, T, 0, stream>>>(Wv, Wvt, kHID, kHKV * kD); }
  { int n = kHID * (kH * kD);
    transpose_bf16_kernel<<<(n + T - 1) / T, T, 0, stream>>>(Wo, Wot, kH * kD, kHID); }

  // 2) QKV projection GEMMs (TDM double-buffered WMMA)
  gemm_bf16_tn<<<dim3((kH * kD) / 128, kM / 128), T, 0, stream>>>(
      Xbf, Wqt, qf32, kM, kH * kD, kHID);
  gemm_bf16_tn<<<dim3((kHKV * kD) / 128, kM / 128), T, 0, stream>>>(
      Xbf, Wkt, kf32, kM, kHKV * kD, kHID);
  gemm_bf16_tn<<<dim3((kHKV * kD) / 128, kM / 128), T, 0, stream>>>(
      Xbf, Wvt, vf32, kM, kHKV * kD, kHID);

  // 3) RoPE + layout transforms (q pre-scaled; v transposed for P@V fragments)
  { int n = kB * kS * kH * 32;
    rope_q_kernel<<<(n + T - 1) / T, T, 0, stream>>>(qf32, cosT, sinT, qbf); }
  { int n = kB * kS * kHKV * 32;
    rope_kv_kernel<<<(n + T - 1) / T, T, 0, stream>>>(kf32, vf32, cosT, sinT, kbf, vtbf); }

  // 4) causal flash attention (writes bf16 [B,S,H*D]); NOTE attnbf aliases qf32
  flash_attn_kernel<<<dim3(kS / 128, kH, kB), T, 0, stream>>>(qbf, kbf, vtbf, attnbf);

  // 5) output projection -> f32 result
  gemm_bf16_tn<<<dim3(kHID / 128, kM / 128), T, 0, stream>>>(
      attnbf, Wot, out, kM, kHID, kH * kD);
}
